// MultiHeadAttention_3650722202113
// MI455X (gfx1250) — matrix operand order repaired and checked by reading, not compiled, not run on hardware
//
#include <hip/hip_runtime.h>

typedef __attribute__((ext_vector_type(16))) _Float16 v16h;
typedef __attribute__((ext_vector_type(8)))  _Float16 v8h;
typedef __attribute__((ext_vector_type(8)))  float    v8f;
typedef __attribute__((ext_vector_type(4)))  float    v4f;

#define D_MODEL 1024
#define SEQ     2048
#define NHEADS  16
#define DK      64
#define LDP     40   // padded LDS row stride (f16): 80B, 16B-aligned, conflict-free
#define LDA     72   // padded stride for 64-wide tiles: 144B, 16B-aligned, conflict-free

#define NEG_INF (-__builtin_inff())

#define WMMA_F16(A, B, C) \
  __builtin_amdgcn_wmma_f32_16x16x32_f16(false, (A), false, (B), (short)0, (C), false, false)

// ---- WMMA fragment loaders (per CDNA5 ISA 7.12.2 layouts) --------------------
// A 16x32 f16: lane&15 = row M; half-wave selects K in {0..7,16..23} / {8..15,24..31}
__device__ __forceinline__ v16h frag_a(const _Float16* base, int lane, int stride) {
  const int r  = lane & 15;
  const int kh = lane >> 4;
  const _Float16* p = base + r * stride + kh * 8;
  v8h c0 = *(const v8h*)p;
  v8h c1 = *(const v8h*)(p + 16);
  v16h o;
#pragma unroll
  for (int i = 0; i < 8; ++i) { o[i] = c0[i]; o[i + 8] = c1[i]; }
  return o;
}
// B 32x16 f16 from B^T layout [n][k]: lane&15 = col N; half-wave selects K in {0..7,16..23} / {8..15,24..31}
__device__ __forceinline__ v16h frag_b(const _Float16* base, int lane, int stride) {
  const int n  = lane & 15;
  const int kh = lane >> 4;
  const _Float16* p = base + n * stride + kh * 8;
  v8h c0 = *(const v8h*)p;
  v8h c1 = *(const v8h*)(p + 16);
  v16h o;
#pragma unroll
  for (int i = 0; i < 8; ++i) { o[i] = c0[i]; o[i + 8] = c1[i]; }
  return o;
}
// stage 16 consecutive f32 -> 16 f16 (two aligned 16B LDS stores)
__device__ __forceinline__ void stage_f32(const float* src, _Float16* dst) {
  v4f f0 = *(const v4f*)(src);
  v4f f1 = *(const v4f*)(src + 4);
  v4f f2 = *(const v4f*)(src + 8);
  v4f f3 = *(const v4f*)(src + 12);
  v8h h0, h1;
#pragma unroll
  for (int e = 0; e < 4; ++e) {
    h0[e]     = (_Float16)f0[e]; h0[e + 4] = (_Float16)f1[e];
    h1[e]     = (_Float16)f2[e]; h1[e + 4] = (_Float16)f3[e];
  }
  *(v8h*)dst       = h0;
  *(v8h*)(dst + 8) = h1;
}

// ---- 1) projection: C[m,n] = X[m,:] . W[n,:]  -> f16, head-split layout ------
__global__ __launch_bounds__(128) void proj_kernel(const float* __restrict__ X,
                                                   const float* __restrict__ W,
                                                   _Float16* __restrict__ outh) {
  __shared__ _Float16 As[64 * LDP];
  __shared__ _Float16 Bs[64 * LDP];
  const int tid = threadIdx.x, lane = tid & 31, wave = tid >> 5;
  const int wm = (wave >> 1) * 32, wn = (wave & 1) * 32;
  const int m0 = blockIdx.y * 64, n0 = blockIdx.x * 64;
  const int srow = tid >> 1, scol = (tid & 1) * 16;

  v8f acc[2][2] = {};
  for (int k0 = 0; k0 < D_MODEL; k0 += 32) {
    stage_f32(X + (size_t)(m0 + srow) * D_MODEL + k0 + scol, &As[srow * LDP + scol]);
    stage_f32(W + (size_t)(n0 + srow) * D_MODEL + k0 + scol, &Bs[srow * LDP + scol]);
    __syncthreads();
    v16h a0 = frag_a(&As[(wm)*LDP], lane, LDP);
    v16h a1 = frag_a(&As[(wm + 16) * LDP], lane, LDP);
    v16h b0 = frag_b(&Bs[(wn)*LDP], lane, LDP);
    v16h b1 = frag_b(&Bs[(wn + 16) * LDP], lane, LDP);
    acc[0][0] = WMMA_F16(a0, b0, acc[0][0]);
    acc[0][1] = WMMA_F16(a0, b1, acc[0][1]);
    acc[1][0] = WMMA_F16(a1, b0, acc[1][0]);
    acc[1][1] = WMMA_F16(a1, b1, acc[1][1]);
    __syncthreads();
  }
  const int cn = lane & 15, mAdd = (lane >> 4) * 8;
#pragma unroll
  for (int tm = 0; tm < 2; ++tm)
#pragma unroll
    for (int tn = 0; tn < 2; ++tn)
#pragma unroll
      for (int rr = 0; rr < 8; ++rr) {
        const int m = m0 + wm + tm * 16 + mAdd + rr;
        const int n = n0 + wn + tn * 16 + cn;
        const int b = m >> 11, s = m & (SEQ - 1);
        const int h = n >> 6,  d = n & (DK - 1);
        outh[(((size_t)(b * NHEADS + h) * SEQ + s) << 6) + d] = (_Float16)acc[tm][tn][rr];
      }
}

// ---- 2) one-shot v transpose: vh[bh][s][d] -> vT[bh][d][s] -------------------
// Makes A.V B-fragments directly loadable from global, removing all barriers
// and per-i-tile LDS restaging from the flash kernel.
__global__ __launch_bounds__(128) void transpose_v_kernel(const _Float16* __restrict__ vh,
                                                          _Float16* __restrict__ vT) {
  __shared__ _Float16 T[64 * LDA];
  const int tid = threadIdx.x;
  const int j0 = blockIdx.x * 64;
  const int bh = blockIdx.y;
  const _Float16* src = vh + (size_t)bh * SEQ * DK;
  _Float16* dst = vT + (size_t)bh * DK * SEQ;
  {  // coalesced load of 64x64 tile
    const int j = tid >> 1, dB = (tid & 1) * 32;
    const _Float16* p = src + (size_t)(j0 + j) * DK + dB;
    _Float16* t = &T[j * LDA + dB];
    *(v8h*)t        = *(const v8h*)p;
    *(v8h*)(t + 8)  = *(const v8h*)(p + 8);
    *(v8h*)(t + 16) = *(const v8h*)(p + 16);
    *(v8h*)(t + 24) = *(const v8h*)(p + 24);
  }
  __syncthreads();
  {  // transposed coalesced store
    const int d = tid >> 1, jB = (tid & 1) * 32;
    v8h o[4];
#pragma unroll
    for (int c = 0; c < 4; ++c)
#pragma unroll
      for (int e = 0; e < 8; ++e) o[c][e] = T[(jB + c * 8 + e) * LDA + d];
    _Float16* p = dst + (size_t)d * SEQ + j0 + jB;
    *(v8h*)p        = o[0];
    *(v8h*)(p + 8)  = o[1];
    *(v8h*)(p + 16) = o[2];
    *(v8h*)(p + 24) = o[3];
  }
}

// ---- 3) fused: scores + online softmax + attn write + attn.v -----------------
// One workgroup (4 waves) owns a 64-row strip of one (b,h); wave w owns 16 rows.
// Pass 1: QK^T tiles (WMMA), online row max / exp-sum (scores discarded).
// Pass 2: recompute, normalize, write attn (its single HBM pass), WMMA attn.vT.
// Barrier-free: only LDS is the wave-private attn-fragment staging region.
__global__ __launch_bounds__(128) void flash_kernel(const _Float16* __restrict__ qh,
                                                    const _Float16* __restrict__ kh,
                                                    const _Float16* __restrict__ vT,
                                                    float* __restrict__ attn,
                                                    _Float16* __restrict__ oh) {
  __shared__ _Float16 As[64 * LDA];  // 4 wave-private 16-row regions
  const int tid = threadIdx.x, lane = tid & 31, wave = tid >> 5;
  const int w16 = wave * 16;
  const int i0 = blockIdx.x * 64, bh = blockIdx.y;
  const _Float16* q  = qh + (size_t)bh * SEQ * DK;
  const _Float16* k  = kh + (size_t)bh * SEQ * DK;
  const _Float16* vt = vT + (size_t)bh * DK * SEQ;
  float* P = attn + (size_t)bh * SEQ * SEQ + (size_t)i0 * SEQ;
  _Float16* out = oh + (size_t)bh * SEQ * DK + (size_t)i0 * DK;
  const int cn = lane & 15, mAdd = (lane >> 4) * 8;

  // zero-fill attn columns beyond the diagonal block (reference attn == 0 there)
  {
    v4f z = {0.f, 0.f, 0.f, 0.f};
    for (int rr = 0; rr < 16; ++rr) {
      float* prow = P + (size_t)(w16 + rr) * SEQ;
      for (int col = i0 + 64 + lane * 4; col < SEQ; col += 128) *(v4f*)(prow + col) = z;
    }
  }

  // q fragments for this wave's rows, K = 0..63 (L2-resident reads)
  const v16h aq0 = frag_a(q + (size_t)(i0 + w16) * DK, lane, DK);
  const v16h aq1 = frag_a(q + (size_t)(i0 + w16) * DK + 32, lane, DK);

  float rmax[8], rsum[8];
#pragma unroll
  for (int rr = 0; rr < 8; ++rr) { rmax[rr] = NEG_INF; rsum[rr] = 0.f; }

  // ---------------- pass 1: online row statistics ----------------
  for (int j0 = 0; j0 <= i0; j0 += 64) {
    if (j0 + 64 <= i0) {  // prefetch next k tile (lane-spread over its rows)
      __builtin_prefetch(k + (size_t)(j0 + 64 + lane) * DK, 0, 1);
      __builtin_prefetch(k + (size_t)(j0 + 96 + lane) * DK, 0, 1);
    }
    v8f c[4];
#pragma unroll
    for (int t = 0; t < 4; ++t) {
      v8f acc = {};
      v16h b0 = frag_b(k + (size_t)(j0 + t * 16) * DK, lane, DK);
      v16h b1 = frag_b(k + (size_t)(j0 + t * 16) * DK + 32, lane, DK);
      acc = WMMA_F16(aq0, b0, acc);
      acc = WMMA_F16(aq1, b1, acc);
      c[t] = acc;
    }
#pragma unroll
    for (int rr = 0; rr < 8; ++rr) {
      const int irow = i0 + w16 + mAdd + rr;
      float vals[4], mt = NEG_INF;
#pragma unroll
      for (int t = 0; t < 4; ++t) {
        const int j = j0 + t * 16 + cn;
        const float s = (j > irow) ? NEG_INF : c[t][rr] * 0.125f;
        vals[t] = s;
        mt = fmaxf(mt, s);
      }
#pragma unroll
      for (int m = 8; m > 0; m >>= 1) mt = fmaxf(mt, __shfl_xor(mt, m, 32));
      const float mn = fmaxf(rmax[rr], mt);
      float st = 0.f;
#pragma unroll
      for (int t = 0; t < 4; ++t) st += __expf(vals[t] - mn);
#pragma unroll
      for (int m = 8; m > 0; m >>= 1) st += __shfl_xor(st, m, 32);
      rsum[rr] = rsum[rr] * __expf(rmax[rr] - mn) + st;
      rmax[rr] = mn;
    }
  }
  float inv[8];
#pragma unroll
  for (int rr = 0; rr < 8; ++rr) inv[rr] = 1.0f / rsum[rr];

  // ---------- pass 2: normalize, write attn, accumulate attn.vT ---------------
  v8f oacc[4] = {};
  _Float16* Asw = &As[w16 * LDA];  // this wave's private region (LDS in-order per wave)
  for (int j0 = 0; j0 <= i0; j0 += 64) {
    if (j0 + 64 <= i0) {
      __builtin_prefetch(k + (size_t)(j0 + 64 + lane) * DK, 0, 1);
      __builtin_prefetch(k + (size_t)(j0 + 96 + lane) * DK, 0, 1);
    }
#pragma unroll
    for (int t = 0; t < 4; ++t) {
      v8f acc = {};
      v16h b0 = frag_b(k + (size_t)(j0 + t * 16) * DK, lane, DK);
      v16h b1 = frag_b(k + (size_t)(j0 + t * 16) * DK + 32, lane, DK);
      acc = WMMA_F16(aq0, b0, acc);
      acc = WMMA_F16(aq1, b1, acc);
#pragma unroll
      for (int rr = 0; rr < 8; ++rr) {
        const int lrow = w16 + mAdd + rr;
        const int j = j0 + t * 16 + cn;
        const float a =
            (j > i0 + lrow) ? 0.f : __expf(acc[rr] * 0.125f - rmax[rr]) * inv[rr];
        P[(size_t)lrow * SEQ + j] = a;
        Asw[(mAdd + rr) * LDA + t * 16 + cn] = (_Float16)a;
      }
    }
#pragma unroll
    for (int kk2 = 0; kk2 < 2; ++kk2) {
      v16h fa = frag_a(Asw + kk2 * 32, lane, LDA);
#pragma unroll
      for (int t = 0; t < 4; ++t) {
        v16h fb = frag_b(vt + (size_t)(t * 16) * SEQ + j0 + kk2 * 32, lane, SEQ);
        oacc[t] = WMMA_F16(fa, fb, oacc[t]);
      }
    }
  }
#pragma unroll
  for (int t = 0; t < 4; ++t)
#pragma unroll
    for (int rr = 0; rr < 8; ++rr)
      out[(size_t)(w16 + mAdd + rr) * DK + t * 16 + cn] = (_Float16)oacc[t][rr];
}

// ---- 4) out = concat_heads(out_h) . W_o^T + b_o ------------------------------
__global__ __launch_bounds__(128) void oproj_kernel(const _Float16* __restrict__ oh,
                                                    const float* __restrict__ Wo,
                                                    const float* __restrict__ bo,
                                                    float* __restrict__ out) {
  __shared__ _Float16 As[64 * LDP];
  __shared__ _Float16 Bs[64 * LDP];
  const int tid = threadIdx.x, lane = tid & 31, wave = tid >> 5;
  const int wm = (wave >> 1) * 32, wn = (wave & 1) * 32;
  const int m0 = blockIdx.y * 64, n0 = blockIdx.x * 64;
  const int srow = tid >> 1, scol = (tid & 1) * 16;
  const int ms = m0 + srow, bb = ms >> 11, s = ms & (SEQ - 1);
  v8f acc[2][2] = {};
  for (int k0 = 0; k0 < D_MODEL; k0 += 32) {
    {  // A: concat-heads gather (contiguous inside a head's 64-block)
      const _Float16* src =
          oh + (((size_t)(bb * NHEADS + (k0 >> 6)) * SEQ + s) << 6) + (k0 & 63) + scol;
      *(v8h*)&As[srow * LDP + scol]     = *(const v8h*)src;
      *(v8h*)&As[srow * LDP + scol + 8] = *(const v8h*)(src + 8);
    }
    stage_f32(Wo + (size_t)(n0 + srow) * D_MODEL + k0 + scol, &Bs[srow * LDP + scol]);
    __syncthreads();
    v16h a0 = frag_a(&As[(wm)*LDP], lane, LDP);
    v16h a1 = frag_a(&As[(wm + 16) * LDP], lane, LDP);
    v16h b0 = frag_b(&Bs[(wn)*LDP], lane, LDP);
    v16h b1 = frag_b(&Bs[(wn + 16) * LDP], lane, LDP);
    acc[0][0] = WMMA_F16(a0, b0, acc[0][0]);
    acc[0][1] = WMMA_F16(a0, b1, acc[0][1]);
    acc[1][0] = WMMA_F16(a1, b0, acc[1][0]);
    acc[1][1] = WMMA_F16(a1, b1, acc[1][1]);
    __syncthreads();
  }
  const int cn = lane & 15, mAdd = (lane >> 4) * 8;
#pragma unroll
  for (int tm = 0; tm < 2; ++tm)
#pragma unroll
    for (int tn = 0; tn < 2; ++tn)
#pragma unroll
      for (int rr = 0; rr < 8; ++rr) {
        const int m = m0 + wm + tm * 16 + mAdd + rr;
        const int n = n0 + wn + tn * 16 + cn;
        out[(size_t)m * D_MODEL + n] = acc[tm][tn][rr] + bo[n];
      }
}

extern "C" void kernel_launch(void* const* d_in, const int* in_sizes, int n_in,
                              void* d_out, int out_size, void* d_ws, size_t ws_size,
                              hipStream_t stream) {
  const float* Q  = (const float*)d_in[0];
  const float* K  = (const float*)d_in[1];
  const float* V  = (const float*)d_in[2];
  const float* Wq = (const float*)d_in[3];
  const float* Wk = (const float*)d_in[4];
  const float* Wv = (const float*)d_in[5];
  const float* Wo = (const float*)d_in[6];
  const float* bo = (const float*)d_in[7];

  float* out  = (float*)d_out;
  float* attn = out + (size_t)2 * SEQ * D_MODEL;  // second output, flat after `out`

  const size_t perBuf = (size_t)2 * NHEADS * SEQ * DK;  // 4,194,304 f16 elems (8 MB)
  _Float16* qh = (_Float16*)d_ws;
  _Float16* kh = qh + perBuf;
  _Float16* vh = kh + perBuf;
  _Float16* vT = vh + perBuf;
  _Float16* oh = vT + perBuf;  // total ws use: 40 MB

  const dim3 blk(128);
  proj_kernel<<<dim3(16, 64), blk, 0, stream>>>(Q, Wq, qh);
  proj_kernel<<<dim3(16, 64), blk, 0, stream>>>(K, Wk, kh);
  proj_kernel<<<dim3(16, 64), blk, 0, stream>>>(V, Wv, vh);
  transpose_v_kernel<<<dim3(32, 32), blk, 0, stream>>>(vh, vT);
  flash_kernel<<<dim3(32, 32), blk, 0, stream>>>(qh, kh, vT, attn, oh);
  oproj_kernel<<<dim3(16, 64), blk, 0, stream>>>(oh, Wo, bo, out);
}
